// Baseline_LatentSet_Single_46901042872714
// MI455X (gfx1250) — matrix operand order repaired and checked by reading, not compiled, not run on hardware
//
#include <hip/hip_runtime.h>
#include <hip/hip_bf16.h>
#include <math.h>

typedef __attribute__((ext_vector_type(16))) _Float16 v16h;
typedef __attribute__((ext_vector_type(8)))  _Float16 v8h;
typedef __attribute__((ext_vector_type(8)))  float    v8f;

#define G_   16
#define K_   4
#define C_   256
#define H_   64
#define W_   44
#define SH_  4
#define P_   176      // SH_*W_
#define N_   64
#define KK_  22       // int(176*0.125)
#define CP_  264      // padded stripe row stride in halves (528B, 16B-aligned, bank-skewed)
#define PW_  46       // padded halo width  (W_+2)
#define PH_  6        // padded halo height (SH_+2)
#define PSZ_ (PH_*PW_)  // 276
#define TEMP_INV 8.0f
#define GATE_THR 0.05f
#define EPS_ 1e-6f

__device__ __forceinline__ v16h ld16h(const _Float16* p0, const _Float16* p1) {
  v8h lo = *(const v8h*)p0;   // ds_load_b128
  v8h hi = *(const v8h*)p1;   // ds_load_b128
  return __builtin_shufflevector(lo, hi, 0,1,2,3,4,5,6,7,8,9,10,11,12,13,14,15);
}

// ---------------- Kernel A: l2-normalize latent basis -> ws ----------------
__global__ void normalize_basis_k(const float* __restrict__ lb, float* __restrict__ lbn) {
  __shared__ float red[256];
  int gk = blockIdx.x;          // 0..63  (g*K+k)
  int c  = threadIdx.x;         // 0..255
  float v = lb[(size_t)gk * C_ + c];
  red[c] = v * v;
  __syncthreads();
  for (int off = 128; off > 0; off >>= 1) {
    if (c < off) red[c] += red[c + off];
    __syncthreads();
  }
  float nrm = sqrtf(red[0]);
  lbn[(size_t)gk * C_ + c] = v / fmaxf(nrm, 1e-12f);
}

// ---------------- Kernel C: per-(n,g) stripe pipeline ----------------
__global__ __launch_bounds__(256)
void stripe_kernel(const float* __restrict__ x, const float* __restrict__ lbn,
                   float* __restrict__ outTok, float* __restrict__ outS,
                   float* __restrict__ outP, float* __restrict__ wsPres) {
  extern __shared__ char dynsm[];
  _Float16* stripeP = (_Float16*)dynsm;        // [P_][CP_]  row-major f16 stripe (A operand)
  _Float16* basisN  = stripeP + P_ * CP_;      // [16][C_]   f16 basis rows (B operand), rows>=4 zero

  __shared__ float csb[4][PSZ_];      // 4 channels, zero-padded halo (borders stay 0)
  __shared__ float red[256];          // block reductions
  __shared__ float rdPad[PSZ_ * K_];  // rawdot with zero halo
  __shared__ float suppL[P_ * K_];    // support
  __shared__ float pwL[P_ * K_];      // pool weights
  __shared__ float ksum[K_];

  const int tid  = threadIdx.x;
  const int lane = tid & 31;
  const int wave = tid >> 5;
  const int n = blockIdx.x / G_;
  const int g = blockIdx.x % G_;
  const int s  = tid / W_;            // valid when tid < P_
  const int ww = tid % W_;

  // --- stage normalized basis as f16 rows, zero-padded to 16 rows ---
  for (int i = tid; i < 16 * C_; i += 256) {
    int nn = i >> 8, c = i & 255;
    float bv = (nn < K_) ? lbn[((size_t)g * K_ + nn) * C_ + c] : 0.0f;
    basisN[nn * C_ + c] = (_Float16)bv;
  }
  // zero halo buffers
  for (int i = tid; i < 4 * PSZ_; i += 256) ((float*)csb)[i] = 0.f;
  __syncthreads();

  // --- phase 1: stream channels (4 per barrier pair); fe, pooled-norm^2, f16 stage ---
  const float* xg = x + (size_t)n * C_ * H_ * W_ + (size_t)g * SH_ * W_;
  float fe = 0.f, nrm2 = 0.f;
  for (int ch = 0; ch < C_; ch += 4) {
    float vbuf[4];
    if (tid < P_) {
#pragma unroll
      for (int q = 0; q < 4; ++q) {
        float v = xg[(size_t)(ch + q) * H_ * W_ + tid];
        vbuf[q] = v;
        csb[q][(s + 1) * PW_ + (ww + 1)] = v;
        stripeP[tid * CP_ + ch + q] = (_Float16)v;
      }
      if (ch + 4 < C_) __builtin_prefetch(&xg[(size_t)(ch + 4) * H_ * W_ + tid], 0, 0);
    }
    __syncthreads();
    if (tid < P_) {
      const int base = s * PW_ + ww;   // top-left of 3x3 window in padded coords
#pragma unroll
      for (int q = 0; q < 4; ++q) {
        const float* cc = csb[q];
        float acc = cc[base]          + cc[base + 1]        + cc[base + 2]
                  + cc[base + PW_]    + cc[base + PW_ + 1]  + cc[base + PW_ + 2]
                  + cc[base + 2*PW_]  + cc[base + 2*PW_ + 1]+ cc[base + 2*PW_ + 2];
        float pv = acc * (1.0f / 9.0f);       // count_include_pad
        nrm2 += pv * pv;
        fe   += vbuf[q] * vbuf[q];
      }
    }
    __syncthreads();
  }
  fe *= (1.0f / C_);

  // --- active gate: fe / max(fe) > thr, with fallback fe>0 ---
  red[tid] = (tid < P_) ? fe : 0.f;
  __syncthreads();
  for (int off = 128; off > 0; off >>= 1) {
    if (tid < off) red[tid] = fmaxf(red[tid], red[tid + off]);
    __syncthreads();
  }
  float femax = red[0];
  __syncthreads();
  float act = ((fe / fmaxf(femax, EPS_)) > GATE_THR) ? 1.f : 0.f;
  red[tid] = (tid < P_) ? act : 0.f;
  __syncthreads();
  for (int off = 128; off > 0; off >>= 1) {
    if (tid < off) red[tid] += red[tid + off];
    __syncthreads();
  }
  float cnt = red[0];
  __syncthreads();
  if (cnt <= 0.f) act = (fe > 0.f) ? 1.f : 0.f;

  // zero rawdot halo buffer before WMMA writes interior
  for (int i = tid; i < PSZ_ * K_; i += 256) rdPad[i] = 0.f;
  __syncthreads();

  // --- WMMA: rawdot[p][k] = sum_c stripe[p][c] * lbn[k][c] ---
  // M=176 (11 tiles of 16 rows), N=16 (4 used), K=256 (8 steps of 32)
  {
    const int m   = lane & 15;
    const int hiA = (lane >= 16) ? 8  : 0;   // 16-bit A 16x32 layout
    const int hiB = (lane >= 16) ? 16 : 0;   // unused; B is loaded with hiA, same k order as A
    for (int mt = wave; mt < 11; mt += 8) {
      v8f acc = {};
      const _Float16* arow = &stripeP[(mt * 16 + m) * CP_];
      const _Float16* brow = &basisN[m * C_];
#pragma unroll
      for (int cb = 0; cb < C_; cb += 32) {
        v16h a = ld16h(arow + cb + hiA, arow + cb + hiA + 16);
        v16h b = ld16h(brow + cb + hiA, brow + cb + hiA + 16);
        acc = __builtin_amdgcn_wmma_f32_16x16x32_f16(
            false, a, false, b, (short)0, acc, false, false);
      }
      if (m < K_) {
#pragma unroll
        for (int r = 0; r < 8; ++r) {
          int p  = mt * 16 + r + hiA;        // C/D layout: VGPR r -> M=r (+8 for hi lanes)
          int sp = p / W_, wp = p % W_;
          rdPad[((sp + 1) * PW_ + (wp + 1)) * K_ + m] = acc[r];
        }
      }
    }
  }
  __syncthreads();

  // --- pooled dot -> logits -> softmax -> support (branch-free halo pooling) ---
  if (tid < P_) {
    float nrm = fmaxf(sqrtf(nrm2), 1e-12f);
    const int base = (s * PW_ + ww) * K_;    // top-left of 3x3 in padded coords
    float lg[K_];
#pragma unroll
    for (int k = 0; k < K_; ++k) {
      const float* rd = rdPad + k;
      float acc = rd[base]              + rd[base + K_]            + rd[base + 2*K_]
                + rd[base + PW_*K_]     + rd[base + (PW_+1)*K_]    + rd[base + (PW_+2)*K_]
                + rd[base + 2*PW_*K_]   + rd[base + (2*PW_+1)*K_]  + rd[base + (2*PW_+2)*K_];
      lg[k] = (acc * (1.0f / 9.0f)) / nrm * TEMP_INV;  // pooling of dot == dot of pooled
    }
    float mx = fmaxf(fmaxf(lg[0], lg[1]), fmaxf(lg[2], lg[3]));
    float ex[K_], es = 0.f;
#pragma unroll
    for (int k = 0; k < K_; ++k) { ex[k] = __expf(lg[k] - mx); es += ex[k]; }
#pragma unroll
    for (int k = 0; k < K_; ++k) suppL[tid * K_ + k] = (ex[k] / es) * act;
  }
  __syncthreads();

  // --- per-latent sums over p (waves 0..3) ---
  if (wave < K_) {
    int k = wave;
    float sum = 0.f;
    for (int p = lane; p < P_; p += 32) sum += suppL[p * K_ + k];
    for (int off = 16; off > 0; off >>= 1) sum += __shfl_xor(sum, off, 32);
    if (lane == 0) ksum[k] = sum;
  }
  __syncthreads();

  // --- pool weights + scatter into global zero maps ---
  if (tid < P_) {
    size_t base = ((size_t)n * (H_ * W_) + (size_t)g * P_ + tid) * (G_ * K_) + (size_t)g * K_;
#pragma unroll
    for (int k = 0; k < K_; ++k) {
      float sv = suppL[tid * K_ + k];
      float pv = sv / fmaxf(ksum[k], EPS_);
      pwL[tid * K_ + k] = pv;
      outS[base + k] = sv;
      outP[base + k] = pv;
    }
  }
  __syncthreads();

  // --- tokens[k][c] = sum_p pw[p][k] * x[c][p] (f32 re-read; x is L2-resident) ---
  {
    const int c = tid;
    const float4* xc4 = (const float4*)(xg + (size_t)c * H_ * W_);  // 16B-aligned
    float a0 = 0, a1 = 0, a2 = 0, a3 = 0;
    for (int j = 0; j < P_ / 4; ++j) {
      float4 xv = xc4[j];
      const float* pw = &pwL[j * 4 * K_];
      a0 += pw[0] * xv.x + pw[K_+0] * xv.y + pw[2*K_+0] * xv.z + pw[3*K_+0] * xv.w;
      a1 += pw[1] * xv.x + pw[K_+1] * xv.y + pw[2*K_+1] * xv.z + pw[3*K_+1] * xv.w;
      a2 += pw[2] * xv.x + pw[K_+2] * xv.y + pw[2*K_+2] * xv.z + pw[3*K_+2] * xv.w;
      a3 += pw[3] * xv.x + pw[K_+3] * xv.y + pw[2*K_+3] * xv.z + pw[3*K_+3] * xv.w;
    }
    size_t tb = ((size_t)n * C_ + c) * (G_ * K_) + (size_t)g * K_;
    outTok[tb + 0] = a0; outTok[tb + 1] = a1;
    outTok[tb + 2] = a2; outTok[tb + 3] = a3;
  }

  // --- presence: mean of top-22 support per latent (waves 0..3) ---
  if (wave < K_) {
    int k = wave;
    float vals[6];
#pragma unroll
    for (int j = 0; j < 6; ++j) {
      int p = lane + 32 * j;
      vals[j] = (p < P_) ? suppL[p * K_ + k] : -1.0f;   // support >= 0
    }
    float tsum = 0.f;
    for (int it = 0; it < KK_; ++it) {
      float lm = vals[0]; int li = 0;
#pragma unroll
      for (int j = 1; j < 6; ++j) if (vals[j] > lm) { lm = vals[j]; li = j; }
      float bv = lm; int bl = lane;
      for (int off = 16; off > 0; off >>= 1) {
        float ov = __shfl_xor(bv, off, 32);
        int   ol = __shfl_xor(bl, off, 32);
        if (ov > bv || (ov == bv && ol < bl)) { bv = ov; bl = ol; }
      }
      if (lane == bl) vals[li] = -1.0f;   // winner removes its element
      tsum += bv;
    }
    if (lane == 0) wsPres[(size_t)n * (G_ * K_) + g * K_ + k] = tsum / (float)KK_;
  }
}

// ---------------- Kernel D: normalize presence rows ----------------
__global__ void presence_norm_k(const float* __restrict__ wsPres, float* __restrict__ outPres) {
  __shared__ float red[64];
  int n = blockIdx.x, j = threadIdx.x;   // 64 threads
  float v = wsPres[(size_t)n * 64 + j];
  red[j] = v;
  __syncthreads();
  for (int off = 32; off > 0; off >>= 1) {
    if (j < off) red[j] += red[j + off];
    __syncthreads();
  }
  outPres[(size_t)n * 64 + j] = v / fmaxf(red[0], EPS_);
}

extern "C" void kernel_launch(void* const* d_in, const int* in_sizes, int n_in,
                              void* d_out, int out_size, void* d_ws, size_t ws_size,
                              hipStream_t stream) {
  const float* x  = (const float*)d_in[0];  // [64,256,64,44]
  const float* lb = (const float*)d_in[1];  // [16,4,256]
  float* out = (float*)d_out;

  float* outTok  = out;                                        // [64,256,64]
  float* outPres = outTok + (size_t)N_ * C_ * G_ * K_;         // [64,64]
  float* outS    = outPres + (size_t)N_ * G_ * K_;             // [64,2816,64]
  float* outP    = outS + (size_t)N_ * H_ * W_ * G_ * K_;      // [64,2816,64]

  float* lbn    = (float*)d_ws;                                // 16384 f32
  float* wsPres = lbn + (size_t)G_ * K_ * C_;                  // 4096 f32

  // zero everything (scatter maps are mostly zero; tokens/presence overwritten)
  hipMemsetAsync(d_out, 0, (size_t)out_size * sizeof(float), stream);

  normalize_basis_k<<<G_ * K_, 256, 0, stream>>>(lb, lbn);

  size_t dynLds = (size_t)(P_ * CP_ + 16 * C_) * sizeof(_Float16);  // ~101 KB (<320 KB/WGP)
  stripe_kernel<<<N_ * G_, 256, dynLds, stream>>>(x, lbn, outTok, outS, outP, wsPres);

  presence_norm_k<<<N_, 64, 0, stream>>>(wsPres, outPres);
}
